// TFAdaptiveSoftmaxMask_76149770158357
// MI455X (gfx1250) — matrix operand order repaired and checked by reading, not compiled, not run on hardware
//
#include <hip/hip_runtime.h>
#include <hip/hip_bf16.h>

// ---------------------------------------------------------------------------
// Adaptive softmax on MI455X (gfx1250).
// Roofline: 548MB output write dominates (~24us at 23.3TB/s). 37 GFLOP of
// GEMM runs on v_wmma_f32_16x16x32_bf16 (f32 accumulate); all W fit in the
// 192MB L2. W pre-converted to bf16 in scratch when ws_size allows, so the
// GEMM inner loop is 2x b128 A loads + 1 WMMA. 4 M-tiles per wave share each
// B tile (4x less W traffic).
// ---------------------------------------------------------------------------

typedef __attribute__((ext_vector_type(16))) __bf16 v16bf;
typedef __attribute__((ext_vector_type(8)))  __bf16 v8bf;
typedef __attribute__((ext_vector_type(4)))  __bf16 v4bf;
typedef __attribute__((ext_vector_type(8)))  float  v8f;
typedef __attribute__((ext_vector_type(4)))  float  v4f;

#define MROWS 512
#define DPROJ 1024
#define VOCAB 267735
#define OUT_ELEMS 137080320ull   // 512 * 267735

static __device__ __forceinline__ v16bf cat16(v8bf lo, v8bf hi) {
  return __builtin_shufflevector(lo, hi, 0,1,2,3,4,5,6,7,8,9,10,11,12,13,14,15);
}
static __device__ __forceinline__ v8bf cat8(v4bf lo, v4bf hi) {
  return __builtin_shufflevector(lo, hi, 0,1,2,3,4,5,6,7);
}
// 16 contiguous f32 -> v16bf via hardware cvt (v_cvt_pk_bf16_f32)
static __device__ __forceinline__ v16bf cvt16(const float* p) {
  v4bf c0 = __builtin_convertvector(*(const v4f*)(p),      v4bf);
  v4bf c1 = __builtin_convertvector(*(const v4f*)(p + 4),  v4bf);
  v4bf c2 = __builtin_convertvector(*(const v4f*)(p + 8),  v4bf);
  v4bf c3 = __builtin_convertvector(*(const v4f*)(p + 12), v4bf);
  return cat16(cat8(c0, c1), cat8(c2, c3));
}

// -------------------- zero-pad y3 (K=16 padded to Kstride=32) --------------
__global__ void zero_pad_kernel(unsigned short* y3) {
  int i = blockIdx.x * 256 + threadIdx.x;
  if (i < MROWS * 32) y3[i] = 0;
}

// -------------------- f32 -> bf16 bulk conversion (n4 float4 quads) --------
__global__ void cvt_bf16_kernel(const float* __restrict__ src,
                                __bf16* __restrict__ dst, long n4) {
  long i = (long)blockIdx.x * blockDim.x + threadIdx.x;
  long stride = (long)gridDim.x * blockDim.x;
  for (; i < n4; i += stride) {
    v4f f = *(const v4f*)(src + i * 4);
    *(v4bf*)(dst + i * 4) = __builtin_convertvector(f, v4bf);
  }
}

// W3: [67735 x 16] f32 -> [67735 x 32] bf16 with zero K-pad
__global__ void cvt_w3_kernel(const float* __restrict__ src,
                              __bf16* __restrict__ dst) {
  const long total = 67735L * 32;
  long i = (long)blockIdx.x * blockDim.x + threadIdx.x;
  long stride = (long)gridDim.x * blockDim.x;
  for (; i < total; i += stride) {
    int c = (int)(i & 31);
    long row = i >> 5;
    dst[i] = (c < 16) ? (__bf16)src[row * 16 + c] : (__bf16)0.0f;
  }
}

// -------------------- projection GEMM: y = hidden @ proj^T -----------------
// A = hidden [512 x 1024] f32 (cvt), B = proj [ND x 1024] f32 (cvt).
// One 16x16 tile per wave, 8 waves/block. Output bf16 with row stride KSTR.
template <int ND, int KSTR>
__global__ void proj_gemm(const float* __restrict__ hidden,
                          const float* __restrict__ proj,
                          __bf16* __restrict__ y) {
  int w = blockIdx.x * 8 + (threadIdx.x >> 5);
  constexpr int NT = ND / 16;
  if (w >= 32 * NT) return;
  int lane = threadIdx.x & 31;
  int lrow = lane & 15, lhi = lane >> 4;
  int mtile = w & 31, ntile = w >> 5;
  int mBase = mtile * 16, nBase = ntile * 16;
  int m = mBase + lrow;
  int n = nBase + lrow;
  const float* hrow = hidden + (size_t)m * DPROJ;
  const float* prow = proj   + (size_t)n * DPROJ;
  v8f acc = {};
  for (int k0 = 0; k0 < DPROJ; k0 += 32) {
    // A: lane holds row m, K = k0+lhi*16 .. +15
    int ab = k0 + lhi * 16;
    v4bf a0 = __builtin_convertvector(*(const v4f*)(hrow + ab),      v4bf);
    v4bf a1 = __builtin_convertvector(*(const v4f*)(hrow + ab + 4),  v4bf);
    v4bf a2 = __builtin_convertvector(*(const v4f*)(hrow + ab + 8), v4bf);
    v4bf a3 = __builtin_convertvector(*(const v4f*)(hrow + ab + 12), v4bf);
    v16bf a = cat16(cat8(a0, a1), cat8(a2, a3));
    // B: lane holds column n (= row n of proj), K = k0+lhi*16 .. +15
    v16bf b = cvt16(prow + k0 + lhi * 16);
    acc = __builtin_amdgcn_wmma_f32_16x16x32_bf16(false, a, false, b,
                                                  (short)0, acc, false, false);
  }
  // C layout: VGPR r -> (M = mBase + r + 8*lhi, N = nBase + lrow)
#pragma unroll
  for (int r = 0; r < 8; ++r) {
    int mm = mBase + r + 8 * lhi;
    y[(size_t)mm * KSTR + n] = (__bf16)acc[r];
  }
}

// -------------------- cluster logits: y0 @ cluster_weight^T ----------------
__global__ void cluster_logits_kernel(const __bf16* __restrict__ y0,
                                      const float* __restrict__ cw,
                                      const float* __restrict__ cb,
                                      float* __restrict__ clusl) {
  int t = blockIdx.x * 256 + threadIdx.x;
  if (t >= MROWS * 3) return;
  int m = t / 3, j = t - m * 3;
  const __bf16* yr = y0 + (size_t)m * DPROJ;
  const float*  wr = cw + (size_t)j * DPROJ;
  float s = 0.f;
  for (int e = 0; e < DPROJ; ++e) s += (float)yr[e] * wr[e];
  clusl[t] = s + cb[j];
}

// -------------------- logit GEMM: out_seg = y @ W^T + b --------------------
// A = y bf16 [512 x KSTR]; B = W, either pre-converted bf16 (stride KSTR)
// or f32 (stride KDIM, cvt on the fly). Each wave: 4 M-tiles x 1 N-tile,
// B tile loaded once per K-step and reused for 4 WMMAs.
template <int KDIM, int KSTR, int NSEG, int VOFF, bool BF16B>
__global__ void logits_gemm(const __bf16* __restrict__ y,
                            const void* __restrict__ Wv,
                            const float* __restrict__ bias,
                            float* __restrict__ out) {
  int w = blockIdx.x * 8 + (threadIdx.x >> 5);
  constexpr int NT = (NSEG + 15) / 16;
  if (w >= 8 * NT) return;
  int lane = threadIdx.x & 31;
  int lrow = lane & 15, lhi = lane >> 4;
  int mgroup = w & 7;          // 4 M-tiles each: rows mgroup*64 .. +63
  int ntile  = w >> 3;         // 8 waves/block share one ntile (L0 reuse of W)
  int nBase  = ntile * 16;
  int n  = nBase + lrow;
  int nn = (n < NSEG) ? n : (NSEG - 1);

  const __bf16* ybase = y + (size_t)(mgroup * 64 + lrow) * KSTR;
  const __bf16* wrow_bf = BF16B ? ((const __bf16*)Wv) + (size_t)nn * KSTR : nullptr;
  const float*  wrow_f  = BF16B ? nullptr : ((const float*)Wv) + (size_t)nn * KDIM;

  v8f acc[4] = {};
  constexpr int KPAD = (KDIM < 32) ? 32 : KDIM;
  for (int k0 = 0; k0 < KPAD; k0 += 32) {
    int kb = k0 + lhi * 16;
    v16bf b;
    if constexpr (BF16B) {
      // stride KSTR includes the zero K-pad for KDIM=16
      v8bf b0 = *(const v8bf*)(wrow_bf + kb);
      v8bf b1 = *(const v8bf*)(wrow_bf + kb + 8);
      b = cat16(b0, b1);
    } else {
      bool full = (KDIM % 32 == 0) ? true : (kb + 16 <= KDIM);
      if (full) {
        b = cvt16(wrow_f + kb);
        if (KDIM >= 64 && k0 + 32 < KPAD)
          __builtin_prefetch(wrow_f + kb + 32, 0, 0);   // global_prefetch_b8
      } else {
#pragma unroll
        for (int e = 0; e < 16; ++e) b[e] = (__bf16)0.0f;
      }
    }
#pragma unroll
    for (int s = 0; s < 4; ++s) {
      const __bf16* yr = ybase + (size_t)s * 16 * KSTR;
      v8bf a0 = *(const v8bf*)(yr + k0 + lhi * 16);
      v8bf a1 = *(const v8bf*)(yr + k0 + lhi * 16 + 8);
      v16bf a = cat16(a0, a1);
      acc[s] = __builtin_amdgcn_wmma_f32_16x16x32_bf16(false, a, false, b,
                                                       (short)0, acc[s],
                                                       false, false);
    }
  }
  if (n < NSEG) {
    float bv = bias[n];
    size_t col = (size_t)VOFF + n;
#pragma unroll
    for (int s = 0; s < 4; ++s) {
#pragma unroll
      for (int r = 0; r < 8; ++r) {
        int mm = mgroup * 64 + s * 16 + r + 8 * lhi;
        out[(size_t)mm * VOCAB + col] = acc[s][r] + bv;
      }
    }
  }
}

// -------------------- streaming log-sum-exp per (segment,row) --------------
__global__ void lse_kernel(const float* __restrict__ out,
                           const float* __restrict__ clusl,
                           float* __restrict__ lse) {
  __shared__ float smx[256], ssm[256];
  int seg = blockIdx.x >> 9;
  int m   = blockIdx.x & 511;
  const int nseg[4] = {20000, 20000, 160000, 67735};
  const int voff[4] = {0, 20000, 40000, 200000};
  int N = nseg[seg];
  const float* row = out + (size_t)m * VOCAB + voff[seg];
  float mx = -3.0e38f, sm = 0.f;
  for (int i = threadIdx.x; i < N; i += 256) {
    float x = row[i];
    if (x > mx) { sm = sm * __expf(mx - x) + 1.f; mx = x; }
    else        { sm += __expf(x - mx); }
  }
  if (seg == 0 && threadIdx.x < 3) {   // head softmax includes cluster logits
    float x = clusl[m * 3 + threadIdx.x];
    if (x > mx) { sm = sm * __expf(mx - x) + 1.f; mx = x; }
    else        { sm += __expf(x - mx); }
  }
  smx[threadIdx.x] = mx; ssm[threadIdx.x] = sm;
  __syncthreads();
  for (int s = 128; s > 0; s >>= 1) {
    if (threadIdx.x < (unsigned)s) {
      float m1 = smx[threadIdx.x],     s1 = ssm[threadIdx.x];
      float m2 = smx[threadIdx.x + s], s2 = ssm[threadIdx.x + s];
      float nm = m1 > m2 ? m1 : m2;
      smx[threadIdx.x] = nm;
      ssm[threadIdx.x] = s1 * __expf(m1 - nm) + s2 * __expf(m2 - nm);
    }
    __syncthreads();
  }
  if (threadIdx.x == 0) lse[seg * 512 + m] = smx[0] + __logf(ssm[0]);
}

// -------------------- finalize: logit -> joint log-prob (float4) -----------
__global__ void finalize_kernel(float* __restrict__ out,
                                const float* __restrict__ lse,
                                const float* __restrict__ clusl) {
  const size_t n4 = OUT_ELEMS / 4;   // exact: OUT_ELEMS % 4 == 0
  size_t vi     = (size_t)blockIdx.x * blockDim.x + threadIdx.x;
  size_t stride = (size_t)gridDim.x * blockDim.x;
  for (; vi < n4; vi += stride) {
    size_t idx = vi * 4;
    v4f v = *(v4f*)(out + idx);
#pragma unroll
    for (int c = 0; c < 4; ++c) {
      size_t id = idx + c;
      size_t m  = id / VOCAB;
      int col   = (int)(id - m * VOCAB);
      int seg   = (col < 20000) ? 0 : (col < 40000) ? 1 : (col < 200000) ? 2 : 3;
      float x = v[c] - lse[seg * 512 + (int)m];
      if (seg > 0) x += clusl[m * 3 + (seg - 1)] - lse[m];
      v[c] = x;
    }
    *(v4f*)(out + idx) = v;
  }
}

// -------------------- loss: -mean gather of gold log-prob ------------------
__global__ void loss_kernel(const float* __restrict__ out,
                            const int* __restrict__ target,
                            float* __restrict__ loss) {
  __shared__ float red[512];
  int t = threadIdx.x;
  red[t] = out[(size_t)t * VOCAB + target[t]];
  __syncthreads();
  for (int s = 256; s > 0; s >>= 1) {
    if (t < s) red[t] += red[t + s];
    __syncthreads();
  }
  if (t == 0) *loss = -red[0] / 512.0f;
}

// ---------------------------------------------------------------------------
extern "C" void kernel_launch(void* const* d_in, const int* in_sizes, int n_in,
                              void* d_out, int out_size, void* d_ws, size_t ws_size,
                              hipStream_t stream) {
  (void)in_sizes; (void)n_in; (void)out_size;
  const float* hidden = (const float*)d_in[0];
  const int*   target = (const int*)d_in[1];
  const float* cw     = (const float*)d_in[2];
  const float* cb     = (const float*)d_in[3];
  const float* W0 = (const float*)d_in[4];
  const float* b0 = (const float*)d_in[5];
  const float* p0 = (const float*)d_in[6];
  const float* W1 = (const float*)d_in[7];
  const float* b1 = (const float*)d_in[8];
  const float* p1 = (const float*)d_in[9];
  const float* W2 = (const float*)d_in[10];
  const float* b2 = (const float*)d_in[11];
  const float* p2 = (const float*)d_in[12];
  const float* W3 = (const float*)d_in[13];
  const float* b3 = (const float*)d_in[14];
  const float* p3 = (const float*)d_in[15];
  float* out  = (float*)d_out;
  float* loss = out + OUT_ELEMS;

  // Workspace layout (bytes)
  char* ws = (char*)d_ws;
  __bf16* y0    = (__bf16*)(ws);             // 512*1024 bf16
  __bf16* y1    = (__bf16*)(ws + 1048576);   // 512*256  bf16
  __bf16* y2    = (__bf16*)(ws + 1310720);   // 512*64   bf16
  __bf16* y3    = (__bf16*)(ws + 1376256);   // 512*32   bf16 (K padded)
  float*  clusl = (float*)(ws + 1409024);    // 512*3 f32
  float*  lse   = (float*)(ws + 1415168);    // 4*512 f32
  // Optional bf16 weight cache (used when scratch is large enough)
  __bf16* W0b = (__bf16*)(ws + 1441792);     // 20,480,000 elem
  __bf16* W1b = (__bf16*)(ws + 42401792);    //  5,120,000 elem
  __bf16* W2b = (__bf16*)(ws + 52641792);    // 10,240,000 elem
  __bf16* W3b = (__bf16*)(ws + 73121792);    // 67735*32 elem (K padded)
  const bool big_ws = ws_size >= 77456832ull;

  zero_pad_kernel<<<dim3(64), dim3(256), 0, stream>>>((unsigned short*)y3);

  proj_gemm<1024, 1024><<<dim3(256), dim3(256), 0, stream>>>(hidden, p0, y0);
  proj_gemm<256,  256 ><<<dim3(64),  dim3(256), 0, stream>>>(hidden, p1, y1);
  proj_gemm<64,   64  ><<<dim3(16),  dim3(256), 0, stream>>>(hidden, p2, y2);
  proj_gemm<16,   32  ><<<dim3(4),   dim3(256), 0, stream>>>(hidden, p3, y3);

  cluster_logits_kernel<<<dim3(6), dim3(256), 0, stream>>>(y0, cw, cb, clusl);

  if (big_ws) {
    cvt_bf16_kernel<<<dim3(4096), dim3(256), 0, stream>>>(W0, W0b, 5120000L);
    cvt_bf16_kernel<<<dim3(2048), dim3(256), 0, stream>>>(W1, W1b, 1280000L);
    cvt_bf16_kernel<<<dim3(4096), dim3(256), 0, stream>>>(W2, W2b, 2560000L);
    cvt_w3_kernel  <<<dim3(2048), dim3(256), 0, stream>>>(W3, W3b);
    logits_gemm<1024, 1024, 20000,  0,      true><<<dim3(1250),  dim3(256), 0, stream>>>(y0, W0b, b0, out);
    logits_gemm<256,  256,  20000,  20000,  true><<<dim3(1250),  dim3(256), 0, stream>>>(y1, W1b, b1, out);
    logits_gemm<64,   64,   160000, 40000,  true><<<dim3(10000), dim3(256), 0, stream>>>(y2, W2b, b2, out);
    logits_gemm<16,   32,   67735,  200000, true><<<dim3(4234),  dim3(256), 0, stream>>>(y3, W3b, b3, out);
  } else {
    logits_gemm<1024, 1024, 20000,  0,      false><<<dim3(1250),  dim3(256), 0, stream>>>(y0, W0, b0, out);
    logits_gemm<256,  256,  20000,  20000,  false><<<dim3(1250),  dim3(256), 0, stream>>>(y1, W1, b1, out);
    logits_gemm<64,   64,   160000, 40000,  false><<<dim3(10000), dim3(256), 0, stream>>>(y2, W2, b2, out);
    logits_gemm<16,   32,   67735,  200000, false><<<dim3(4234),  dim3(256), 0, stream>>>(y3, W3, b3, out);
  }

  lse_kernel<<<dim3(2048), dim3(256), 0, stream>>>(out, clusl, lse);
  finalize_kernel<<<dim3(8192), dim3(256), 0, stream>>>(out, lse, clusl);
  loss_kernel<<<dim3(1), dim3(512), 0, stream>>>(out, target, loss);
}
